// RelativeMultiHeadAttn_1786706395838
// MI455X (gfx1250) — compile-verified, matrix operand order checked by reading, not run on hardware
//
#include <hip/hip_runtime.h>
#include <hip/hip_bf16.h>

#define D_MODEL 1024
#define N_HEAD 16
#define HEAD_DIM 64
#define SEQ_L 1024
#define BATCH 2
#define FDIM 192  // extended QK feature dim: 64 + 4*32

typedef __attribute__((ext_vector_type(16))) _Float16 v16h;
typedef __attribute__((ext_vector_type(8)))  _Float16 v8h;
typedef __attribute__((ext_vector_type(8)))  float    v8f;

// ---------------------------------------------------------------------------
// A/B fragment gather for v_wmma_f32_16x16x32_f16, per CDNA5 ISA layout:
// lanes 0-15 : row = lane,    halves K = [0..7] then [16..23]
// lanes 16-31: row = lane-16, halves K = [8..15] then [24..31]
// Source is a row-major (row, K) tile with rowStrideHalves halves per row.
// Two 16-byte loads per lane.
// ---------------------------------------------------------------------------
__device__ __forceinline__ v16h load_frag(const _Float16* p, int rowStrideHalves) {
  const int lane = threadIdx.x & 31;
  const int r    = lane & 15;
  const int kb   = (lane < 16) ? 0 : 8;
  const _Float16* q = p + r * rowStrideHalves + kb;
  v8h lo = *(const v8h*)(q);
  v8h hi = *(const v8h*)(q + 16);
  v16h out;
#pragma unroll
  for (int i = 0; i < 8; ++i) { out[i] = lo[i]; out[8 + i] = hi[i]; }
  return out;
}

__device__ __forceinline__ v8f wmma_f16(v16h a, v16h b, v8f c) {
  // D = A(16x32, f16) x B(32x16, f16) + C(16x16, f32)
  return __builtin_amdgcn_wmma_f32_16x16x32_f16(false, a, false, b, (short)0, c,
                                                false, false);
}

// ---------------------------------------------------------------------------
// CDNA5 async global->LDS copy (VGPR-bypassing, tracked by ASYNCcnt).
// dst VGPR = per-lane LDS byte offset (low 32 bits of generic shared addr,
// per the LDS aperture rule addr[31:0] == LDS_ADDR), src = 64-bit global VA.
// ---------------------------------------------------------------------------
__device__ __forceinline__ void async_ld_b128(const void* lds_dst, const void* gsrc) {
  unsigned l = (unsigned)(unsigned long long)lds_dst;
  asm volatile("global_load_async_to_lds_b128 %0, %1, off"
               :: "v"(l), "v"(gsrc) : "memory");
}
__device__ __forceinline__ void wait_async0() {
  asm volatile("s_wait_asynccnt 0" ::: "memory");
}

// ---------------------------------------------------------------------------
// Kernel 1: qkv[e,t] = sum_d W[e,d] * x[t,d];  M=3072 (e), N=2048 (t), K=1024
// 64x64 block tile, 8 waves, each wave owns two 16x16 output tiles.
// (f32->f16 conversion happens in the staging path, so this one keeps the
// VGPR-mediated copy.)
// ---------------------------------------------------------------------------
__global__ __launch_bounds__(256) void qkv_gemm_kernel(
    const float* __restrict__ x, const float* __restrict__ W,
    float* __restrict__ Qf, float* __restrict__ Kf, float* __restrict__ Vf) {
  __shared__ __align__(16) _Float16 sA[64 * 32];  // W rows (e-major, k-contig)
  __shared__ __align__(16) _Float16 sB[64 * 32];  // x rows (t-major, k-contig)
  const int e0 = blockIdx.y * 64;
  const int t0 = blockIdx.x * 64;
  const int tid  = threadIdx.x;
  const int wave = tid >> 5;
  const int lane = tid & 31;
  const int mi  = wave >> 1;        // row tile 0..3
  const int ni0 = (wave & 1) * 2;   // first of two col tiles
  const int lrow = tid >> 2, lch = (tid & 3) * 8;
  v8f acc0 = {}; v8f acc1 = {};
  for (int k0 = 0; k0 < 1024; k0 += 32) {
    __syncthreads();
    {
      const float* srcA = W + (size_t)(e0 + lrow) * 1024 + k0 + lch;
      const float* srcB = x + (size_t)(t0 + lrow) * 1024 + k0 + lch;
      _Float16* dA = &sA[lrow * 32 + lch];
      _Float16* dB = &sB[lrow * 32 + lch];
#pragma unroll
      for (int i = 0; i < 8; ++i) { dA[i] = (_Float16)srcA[i]; dB[i] = (_Float16)srcB[i]; }
    }
    __syncthreads();
    v16h a  = load_frag(&sA[mi * 16 * 32], 32);
    v16h b0 = load_frag(&sB[(ni0 * 16) * 32], 32);
    v16h b1 = load_frag(&sB[((ni0 + 1) * 16) * 32], 32);
    acc0 = wmma_f16(a, b0, acc0);
    acc1 = wmma_f16(a, b1, acc1);
  }
  const int col = lane & 15;
#pragma unroll
  for (int ti = 0; ti < 2; ++ti) {
    v8f acc = ti ? acc1 : acc0;
    int t = t0 + (ni0 + ti) * 16 + col;
    int b = t >> 10, l = t & 1023;
#pragma unroll
    for (int i = 0; i < 8; ++i) {
      int m = (lane < 16) ? i : (8 + i);
      int e = e0 + mi * 16 + m;
      int typ = e >> 10, rem = e & 1023;
      int n = rem >> 6, d = rem & 63;
      float* dst = (typ == 0) ? Qf : (typ == 1) ? Kf : Vf;
      dst[(((size_t)b * N_HEAD + n) * SEQ_L + l) * HEAD_DIM + d] = acc[i];
    }
  }
}

// ---------------------------------------------------------------------------
// Kernel 2: build 192-dim extended features (angle-addition factorization of
// the relative-position terms) in f16, plus transposed f16 V.
//   Qh = [q+rrb | A_{q+rwb} | B_{q+rwb} | sin(q*w) | cos(q*w)]
//   Kh = [k     | sin(k*w)  | cos(k*w)  | A_k      | B_k     ]
// so  Qh . Kh == (q+rrb).k + (q+rwb).pos[k-q+L] + k.pos[q-k+L]
// ---------------------------------------------------------------------------
__global__ __launch_bounds__(256) void feature_kernel(
    const float* __restrict__ Qf, const float* __restrict__ Kf,
    const float* __restrict__ Vf, const float* __restrict__ rrb,
    const float* __restrict__ rwb, _Float16* __restrict__ Qh,
    _Float16* __restrict__ Kh, _Float16* __restrict__ Vt) {
  int idx = blockIdx.x * blockDim.x + threadIdx.x;  // (b,n,l) flat
  if (idx >= BATCH * N_HEAD * SEQ_L) return;
  int bn = idx >> 10;
  int l  = idx & 1023;
  int n  = bn & (N_HEAD - 1);
  const float* q = Qf + (size_t)idx * HEAD_DIM;
  const float* k = Kf + (size_t)idx * HEAD_DIM;
  const float* v = Vf + (size_t)idx * HEAD_DIM;
  _Float16* qh = Qh + (size_t)idx * FDIM;
  _Float16* kh = Kh + (size_t)idx * FDIM;
  float s[32], c[32];
#pragma unroll
  for (int j = 0; j < 32; ++j) {
    float w = __expf((float)j * (-9.2103403719761836f / 31.0f));  // 1/f^{j/31}
    float a = (float)l * w;
    s[j] = sinf(a); c[j] = cosf(a);
  }
#pragma unroll
  for (int d = 0; d < 64; ++d) qh[d] = (_Float16)(q[d] + rrb[n * 64 + d]);
#pragma unroll
  for (int j = 0; j < 32; ++j) {
    float a0 = q[j]      + rwb[n * 64 + j];
    float a1 = q[32 + j] + rwb[n * 64 + 32 + j];
    qh[64  + j] = (_Float16)( a0 * c[j] + a1 * s[j]);
    qh[96  + j] = (_Float16)(-a0 * s[j] + a1 * c[j]);
    qh[128 + j] = (_Float16)s[j];
    qh[160 + j] = (_Float16)c[j];
  }
#pragma unroll
  for (int d = 0; d < 64; ++d) kh[d] = (_Float16)k[d];
#pragma unroll
  for (int j = 0; j < 32; ++j) {
    kh[64  + j] = (_Float16)s[j];
    kh[96  + j] = (_Float16)c[j];
    kh[128 + j] = (_Float16)( k[j] * c[j] + k[32 + j] * s[j]);
    kh[160 + j] = (_Float16)(-k[j] * s[j] + k[32 + j] * c[j]);
  }
#pragma unroll
  for (int d = 0; d < 64; ++d)
    Vt[((size_t)bn * HEAD_DIM + d) * SEQ_L + l] = (_Float16)v[d];
}

// ---------------------------------------------------------------------------
// Kernel 3: flash attention per (head, 128-row q slab). 8 waves x 16 q rows.
// K/V tiles staged with global_load_async_to_lds_b128 (ASYNCcnt), logits GEMM
// K=192 (6 WMMAs per 16x16 tile), online softmax with half-wave shuffles,
// P.V with K=32 tiles.
// ---------------------------------------------------------------------------
__global__ __launch_bounds__(256) void attn_kernel(
    const _Float16* __restrict__ Qh, const _Float16* __restrict__ Kh,
    const _Float16* __restrict__ Vt, const int* __restrict__ mask,
    float* __restrict__ out) {
  __shared__ __align__(16) _Float16 sK[32 * FDIM];    // 12 KB: 32 keys x 192
  __shared__ __align__(16) _Float16 sV[64 * 32];      //  4 KB: 64 d x 32 keys
  __shared__ __align__(16) _Float16 sP[8][16 * 32];   //  8 KB: per-wave P
  const int wg   = blockIdx.x;
  const int bn   = wg >> 3;   // 32 heads
  const int slab = wg & 7;    // 8 slabs of 128 q rows
  const int b    = bn >> 4;
  const int n    = bn & 15;
  const int tid  = threadIdx.x;
  const int wave = tid >> 5;
  const int lane = tid & 31;
  const int col  = lane & 15;
  const int q0   = slab * 128 + wave * 16;

  // Q-tile A fragments stay resident in VGPRs (6 x v16h = 48 VGPRs)
  const _Float16* Qbase = Qh + ((size_t)bn * SEQ_L + q0) * FDIM;
  v16h qa[6];
#pragma unroll
  for (int cc = 0; cc < 6; ++cc) qa[cc] = load_frag(Qbase + cc * 32, FDIM);

  v8f acc[4] = {};          // 16 q x 64 d output accumulator (f32)
  float rmax[8], rsum[8];   // per-lane: 8 rows of this lane's half
#pragma unroll
  for (int i = 0; i < 8; ++i) { rmax[i] = -1e30f; rsum[i] = 0.f; }

  const int vrow = tid >> 2, vch = (tid & 3) * 8;

  for (int kb = 0; kb < SEQ_L / 32; ++kb) {
    __syncthreads();  // previous iteration's LDS readers done
    {
      const _Float16* gK = Kh + ((size_t)bn * SEQ_L + kb * 32) * FDIM;
      __builtin_prefetch(gK + 32 * FDIM, 0, 0);  // global_prefetch next block
#pragma unroll
      for (int it = 0; it < 3; ++it) {
        int idx = it * 256 + tid;  // 6144 halves, 16B per lane per iter
        async_ld_b128(&sK[idx * 8], gK + idx * 8);
      }
      async_ld_b128(&sV[vrow * 32 + vch],
                    Vt + ((size_t)bn * HEAD_DIM + vrow) * SEQ_L + kb * 32 + vch);
      wait_async0();  // this wave's async LDS writes complete
    }
    __syncthreads();  // all waves' tiles visible

    // ---- logits: S = Qhat . Khat^T over K=192 ----
    v8f s0 = {}, s1 = {};
#pragma unroll
    for (int cc = 0; cc < 6; ++cc)
      s0 = wmma_f16(qa[cc], load_frag(&sK[0 * 16 * FDIM + cc * 32], FDIM), s0);
#pragma unroll
    for (int cc = 0; cc < 6; ++cc)
      s1 = wmma_f16(qa[cc], load_frag(&sK[1 * 16 * FDIM + cc * 32], FDIM), s1);

    // scale (1/sqrt(64)) + mask
    int key0 = kb * 32 + col;
    float mz0 = (mask[b * SEQ_L + key0] == 0) ? -1e30f : 0.0f;
    float mz1 = (mask[b * SEQ_L + key0 + 16] == 0) ? -1e30f : 0.0f;
#pragma unroll
    for (int i = 0; i < 8; ++i) {
      s0[i] = s0[i] * 0.125f + mz0;
      s1[i] = s1[i] * 0.125f + mz1;
    }

    // ---- online softmax (row reductions across each 16-lane half) ----
    float alpha[8];
#pragma unroll
    for (int i = 0; i < 8; ++i) {
      float v = fmaxf(s0[i], s1[i]);
#pragma unroll
      for (int m = 8; m >= 1; m >>= 1) v = fmaxf(v, __shfl_xor(v, m, 32));
      float nm = fmaxf(rmax[i], v);
      alpha[i] = __expf(rmax[i] - nm);
      rmax[i] = nm;
    }
#pragma unroll
    for (int i = 0; i < 8; ++i) {
      s0[i] = __expf(s0[i] - rmax[i]);
      s1[i] = __expf(s1[i] - rmax[i]);
      float rs = s0[i] + s1[i];
#pragma unroll
      for (int m = 8; m >= 1; m >>= 1) rs += __shfl_xor(rs, m, 32);
      rsum[i] = rsum[i] * alpha[i] + rs;
    }
#pragma unroll
    for (int dt = 0; dt < 4; ++dt)
#pragma unroll
      for (int i = 0; i < 8; ++i) acc[dt][i] *= alpha[i];

    // ---- P (f16) via LDS, then acc += P . V  (K=32) ----
    {
      _Float16* p = &sP[wave][0];
#pragma unroll
      for (int i = 0; i < 8; ++i) {
        int r = (lane < 16) ? i : (8 + i);
        p[r * 32 + col]      = (_Float16)s0[i];
        p[r * 32 + 16 + col] = (_Float16)s1[i];
      }
    }
    __syncthreads();
    v16h pa = load_frag(&sP[wave][0], 32);
#pragma unroll
    for (int dt = 0; dt < 4; ++dt) {
      v16h vf = load_frag(&sV[dt * 16 * 32], 32);
      acc[dt] = wmma_f16(pa, vf, acc[dt]);
    }
  }

  // final: out[b][q][n*64+d] = acc / rowsum   (output already transposed form)
#pragma unroll
  for (int dt = 0; dt < 4; ++dt) {
    int d = n * HEAD_DIM + dt * 16 + col;
#pragma unroll
    for (int i = 0; i < 8; ++i) {
      int r = (lane < 16) ? i : (8 + i);
      int q = q0 + r;
      out[((size_t)b * SEQ_L + q) * D_MODEL + d] = acc[dt][i] / rsum[i];
    }
  }
}

// ---------------------------------------------------------------------------
extern "C" void kernel_launch(void* const* d_in, const int* in_sizes, int n_in,
                              void* d_out, int out_size, void* d_ws, size_t ws_size,
                              hipStream_t stream) {
  const float* x    = (const float*)d_in[0];   // (B, L, 1024)
  const int*   mask = (const int*)d_in[1];     // (B, L)
  const float* W    = (const float*)d_in[2];   // (3072, 1024)
  const float* rrb  = (const float*)d_in[3];   // (16, 64)
  const float* rwb  = (const float*)d_in[4];   // (16, 64)
  float* out = (float*)d_out;                  // (B, L, 1024) f32

  char* ws = (char*)d_ws;
  size_t off = 0;
  float* Qf = (float*)(ws + off); off += (size_t)BATCH * N_HEAD * SEQ_L * HEAD_DIM * 4;
  float* Kf = (float*)(ws + off); off += (size_t)BATCH * N_HEAD * SEQ_L * HEAD_DIM * 4;
  float* Vf = (float*)(ws + off); off += (size_t)BATCH * N_HEAD * SEQ_L * HEAD_DIM * 4;
  _Float16* Qh = (_Float16*)(ws + off); off += (size_t)BATCH * N_HEAD * SEQ_L * FDIM * 2;
  _Float16* Kh = (_Float16*)(ws + off); off += (size_t)BATCH * N_HEAD * SEQ_L * FDIM * 2;
  _Float16* Vt = (_Float16*)(ws + off); off += (size_t)BATCH * N_HEAD * HEAD_DIM * SEQ_L * 2;
  // total ws use: 52 MB

  dim3 g1(32, 48);  // t tiles x e tiles
  qkv_gemm_kernel<<<g1, 256, 0, stream>>>(x, W, Qf, Kf, Vf);
  feature_kernel<<<(BATCH * N_HEAD * SEQ_L) / 256, 256, 0, stream>>>(
      Qf, Kf, Vf, rrb, rwb, Qh, Kh, Vt);
  attn_kernel<<<BATCH * N_HEAD * 8, 256, 0, stream>>>(Qh, Kh, Vt, mask, out);
}
